// PhysicsGNN_48447231099240
// MI455X (gfx1250) — compile-verified, matrix operand order checked by reading, not run on hardware
//
#include <hip/hip_runtime.h>

#define HIDDEN 128
#define NNODES 100000
#define NEDGES 1600000
#define NSTEPS 10
#define LN_EPS 1e-5f

typedef __attribute__((ext_vector_type(16))) __bf16 v16bf;
typedef __attribute__((ext_vector_type(8)))  float  v8f;

struct alignas(16) u128 { unsigned int x[4]; };
struct Frag32B { u128 lo, hi; };
union U16x8 { unsigned short u[8]; u128 v; };

__device__ __forceinline__ unsigned short f32_to_bf16(float f) {
  __bf16 h = (__bf16)f;                       // hardware f32->bf16 convert
  return __builtin_bit_cast(unsigned short, h);
}
__device__ __forceinline__ float bf16_to_f32(unsigned short h) {
  return __uint_as_float(((unsigned int)h) << 16);
}

// CDNA5 packed-bf16 atomic add (GLOBAL_ATOMIC_PK_ADD_BF16, no return)
__device__ __forceinline__ void atomic_pk_add_bf16(unsigned short* addr, unsigned int pk) {
  asm volatile("global_atomic_pk_add_bf16 %0, %1, off"
               :: "v"(addr), "v"(pk) : "memory");
}

// CDNA5 async global->LDS 16B copy (tracked by ASYNCcnt)
__device__ __forceinline__ void async_copy16(unsigned lds_off, const void* gaddr) {
  asm volatile("global_load_async_to_lds_b128 %0, %1, off"
               :: "v"(lds_off), "v"(gaddr) : "memory");
}
__device__ __forceinline__ void wait_asynccnt0() {
  asm volatile("s_wait_asynccnt 0" ::: "memory");
}

// Load one 16x32 bf16 WMMA fragment slice for this lane.
// Layout (ISA 7.12.2): lanes 0-15 hold K = [0..7] in elems 0..7 and [16..23] in 8..15;
// lanes 16-31 hold K = [8..15] and [24..31].  base points at row start + kk.
__device__ __forceinline__ v16bf load_frag(const unsigned short* base, int half8) {
  Frag32B f;
  f.lo = *reinterpret_cast<const u128*>(base + half8);
  f.hi = *reinterpret_cast<const u128*>(base + half8 + 16);
  return __builtin_bit_cast(v16bf, f);
}

// 16 rows x 16 cols output tile for one wave: A (16 x K, LDS, row-major bf16)
// times WT (128 x K, bf16, row n = output column n), accumulate f32, bias bv.
template<int K>
__device__ __forceinline__ v8f wmma_rowtile(const unsigned short* __restrict__ A,
                                            const unsigned short* __restrict__ WT,
                                            float bv, int wave, int lane) {
  const int row   = lane & 15;
  const int half8 = (lane >> 4) << 3;
  const unsigned short* arow = A + row * K;
  const unsigned short* brow = WT + (size_t)((wave << 4) + row) * K;
  v8f acc = { bv, bv, bv, bv, bv, bv, bv, bv };
#pragma unroll
  for (int kk = 0; kk < K; kk += 32) {
    v16bf a = load_frag(arow + kk, half8);
    v16bf b = load_frag(brow + kk, half8);
    acc = __builtin_amdgcn_wmma_f32_16x16x32_bf16(false, a, false, b, (short)0, acc,
                                                  false, false);
  }
  return acc;
}

// D layout: elem i -> row M = i + 8*(lane>>4), col = 16*wave + (lane&15)
__device__ __forceinline__ void store_relu_bf16(v8f acc, unsigned short* H, int wave, int lane) {
  const int col   = (wave << 4) + (lane & 15);
  const int mbase = (lane >> 4) << 3;
#pragma unroll
  for (int i = 0; i < 8; ++i)
    H[(mbase + i) * HIDDEN + col] = f32_to_bf16(fmaxf(acc[i], 0.f));
}
__device__ __forceinline__ void store_f32(v8f acc, float* Y, int wave, int lane) {
  const int col   = (wave << 4) + (lane & 15);
  const int mbase = (lane >> 4) << 3;
#pragma unroll
  for (int i = 0; i < 8; ++i)
    Y[(mbase + i) * HIDDEN + col] = acc[i];
}

// Row-wise LayerNorm stats. Thread tid handles row r = tid>>4, slice j = tid&15
// (cols j*8 .. j*8+7). Rows occupy 16-lane halves of a wave32 -> xor masks 1..8.
__device__ __forceinline__ void ln_stats(const float* Yrow, int j, float& mu, float& rs) {
  float s = 0.f, ss = 0.f;
#pragma unroll
  for (int i = 0; i < 8; ++i) { float v = Yrow[j * 8 + i]; s += v; ss += v * v; }
#pragma unroll
  for (int m = 1; m <= 8; m <<= 1) { s += __shfl_xor(s, m, 32); ss += __shfl_xor(ss, m, 32); }
  mu = s * (1.f / 128.f);
  float var = ss * (1.f / 128.f) - mu * mu;
  rs = rsqrtf(var + LN_EPS);
}

// ---------------- prep kernels ----------------

// WT[n*Kp + k] = bf16(W[k*N + n]); zero pad k>=K or n>=N
__global__ __launch_bounds__(256) void wt_conv_kernel(const float* __restrict__ W,
                                                      unsigned short* __restrict__ WT,
                                                      int K, int N, int Kp, int Np) {
  int i = blockIdx.x * 256 + threadIdx.x;
  if (i >= Kp * Np) return;
  int n = i / Kp, k = i - n * Kp;
  float v = (k < K && n < N) ? W[k * N + n] : 0.f;
  WT[i] = f32_to_bf16(v);
}

__global__ __launch_bounds__(256) void zero_f32_kernel(float* __restrict__ p, long long n) {
  long long i = (long long)blockIdx.x * 256 + threadIdx.x;
  if (i < n) p[i] = 0.f;
}

__global__ __launch_bounds__(256) void count_kernel(const int* __restrict__ eidx,
                                                    float* __restrict__ cnt) {
  int i = blockIdx.x * 256 + threadIdx.x;
  unsafeAtomicAdd(&cnt[eidx[NEDGES + i]], 1.f);
}

__global__ __launch_bounds__(256) void inv_clip_kernel(float* __restrict__ cnt, int n) {
  int i = blockIdx.x * 256 + threadIdx.x;
  if (i < n) cnt[i] = 1.f / fmaxf(cnt[i], 1.f);
}

// ---------------- encoder (nodes & edges) ----------------

__global__ __launch_bounds__(256) void encoder_kernel(
    const float* __restrict__ X, int din,
    const unsigned short* __restrict__ WT1, const float* __restrict__ b1,
    const unsigned short* __restrict__ WT2, const float* __restrict__ b2,
    const unsigned short* __restrict__ WT3, const float* __restrict__ b3,
    const float* __restrict__ g, const float* __restrict__ beta,
    unsigned short* __restrict__ out_emb) {
  __shared__ unsigned short Ain[16 * 32];
  __shared__ unsigned short H1[16 * HIDDEN];
  __shared__ unsigned short H2[16 * HIDDEN];
  __shared__ float Y[16 * HIDDEN];
  const int tid = threadIdx.x, wave = tid >> 5, lane = tid & 31;
  const long long tile = blockIdx.x;

  for (int t = tid; t < 16 * 32; t += 256) {
    int r = t >> 5, c = t & 31;
    float v = (c < din) ? X[(tile * 16 + r) * din + c] : 0.f;
    Ain[t] = f32_to_bf16(v);
  }
  __syncthreads();
  { float bv = b1[(wave << 4) + (lane & 15)];
    store_relu_bf16(wmma_rowtile<32>(Ain, WT1, bv, wave, lane), H1, wave, lane); }
  __syncthreads();
  { float bv = b2[(wave << 4) + (lane & 15)];
    store_relu_bf16(wmma_rowtile<HIDDEN>(H1, WT2, bv, wave, lane), H2, wave, lane); }
  __syncthreads();
  { float bv = b3[(wave << 4) + (lane & 15)];
    store_f32(wmma_rowtile<HIDDEN>(H2, WT3, bv, wave, lane), Y, wave, lane); }
  __syncthreads();
  const int r = tid >> 4, j = tid & 15;
  float mu, rs; ln_stats(&Y[r * HIDDEN], j, mu, rs);
  U16x8 nw;
#pragma unroll
  for (int i = 0; i < 8; ++i) {
    int c = j * 8 + i;
    nw.u[i] = f32_to_bf16((Y[r * HIDDEN + c] - mu) * rs * g[c] + beta[c]);
  }
  *reinterpret_cast<u128*>(out_emb + (tile * 16 + r) * HIDDEN + j * 8) = nw.v;
}

// ---------------- edge step: async gather + MLP(384->..)+LN+res + pk-bf16 scatter ----------------

__global__ __launch_bounds__(256) void edge_step_kernel(
    const int* __restrict__ eidx,
    const unsigned short* __restrict__ node_emb,
    unsigned short* __restrict__ edge_emb,
    const unsigned short* __restrict__ WT1, const float* __restrict__ b1,
    const unsigned short* __restrict__ WT2, const float* __restrict__ b2,
    const unsigned short* __restrict__ WT3, const float* __restrict__ b3,
    const float* __restrict__ g, const float* __restrict__ beta,
    unsigned short* __restrict__ agg) {
  __shared__ unsigned short Ain[16 * 384];
  __shared__ unsigned short H1[16 * HIDDEN];
  __shared__ unsigned short H2[16 * HIDDEN];
  __shared__ float Y[16 * HIDDEN];
  const int tid = threadIdx.x, wave = tid >> 5, lane = tid & 31;
  const long long tile = blockIdx.x;
  {
    const int e = tid >> 4, q = tid & 15;
    const long long eg = tile * 16 + e;
    const int s = eidx[eg];
    const int d = eidx[NEDGES + eg];
    // per-lane LDS byte offset = low 32 bits of generic pointer to __shared__
    unsigned abase = (unsigned)(uintptr_t)(Ain + e * 384) + (unsigned)(q * 16);
    async_copy16(abase,       node_emb + (long long)s * HIDDEN + q * 8);
    async_copy16(abase + 256, node_emb + (long long)d * HIDDEN + q * 8);
    async_copy16(abase + 512, edge_emb + eg * HIDDEN + q * 8);
  }
  wait_asynccnt0();
  __syncthreads();
  { float bv = b1[(wave << 4) + (lane & 15)];
    store_relu_bf16(wmma_rowtile<384>(Ain, WT1, bv, wave, lane), H1, wave, lane); }
  __syncthreads();
  { float bv = b2[(wave << 4) + (lane & 15)];
    store_relu_bf16(wmma_rowtile<HIDDEN>(H1, WT2, bv, wave, lane), H2, wave, lane); }
  __syncthreads();
  { float bv = b3[(wave << 4) + (lane & 15)];
    store_f32(wmma_rowtile<HIDDEN>(H2, WT3, bv, wave, lane), Y, wave, lane); }
  __syncthreads();
  const int r = tid >> 4, j = tid & 15;
  const long long eg = tile * 16 + r;
  const int d = eidx[NEDGES + eg];
  float mu, rs; ln_stats(&Y[r * HIDDEN], j, mu, rs);
  U16x8 oldv, nw;
  oldv.v = *reinterpret_cast<const u128*>(edge_emb + eg * HIDDEN + j * 8);
  unsigned short* aggrow = agg + (size_t)d * HIDDEN;
#pragma unroll
  for (int t2 = 0; t2 < 4; ++t2) {
    int c = j * 8 + t2 * 2;
    float v0 = (Y[r * HIDDEN + c]     - mu) * rs * g[c]     + beta[c]     + bf16_to_f32(oldv.u[t2 * 2]);
    float v1 = (Y[r * HIDDEN + c + 1] - mu) * rs * g[c + 1] + beta[c + 1] + bf16_to_f32(oldv.u[t2 * 2 + 1]);
    unsigned short h0 = f32_to_bf16(v0), h1 = f32_to_bf16(v1);
    nw.u[t2 * 2] = h0; nw.u[t2 * 2 + 1] = h1;
    unsigned pk = ((unsigned)h1 << 16) | (unsigned)h0;
    atomic_pk_add_bf16(aggrow + c, pk);   // segment-sum over dst, packed bf16
  }
  *reinterpret_cast<u128*>(edge_emb + eg * HIDDEN + j * 8) = nw.v;
}

// ---------------- node step: [emb | agg*invc] -> MLP(256->..)+LN+res ----------------

__global__ __launch_bounds__(256) void node_step_kernel(
    const unsigned short* __restrict__ agg, const float* __restrict__ invc,
    unsigned short* __restrict__ node_emb,
    const unsigned short* __restrict__ WT1, const float* __restrict__ b1,
    const unsigned short* __restrict__ WT2, const float* __restrict__ b2,
    const unsigned short* __restrict__ WT3, const float* __restrict__ b3,
    const float* __restrict__ g, const float* __restrict__ beta) {
  __shared__ unsigned short Ain[16 * 256];
  __shared__ unsigned short H1[16 * HIDDEN];
  __shared__ unsigned short H2[16 * HIDDEN];
  __shared__ float Y[16 * HIDDEN];
  const int tid = threadIdx.x, wave = tid >> 5, lane = tid & 31;
  const long long tile = blockIdx.x;
  {
    const int e = tid >> 4, q = tid & 15;
    const long long n = tile * 16 + e;
    reinterpret_cast<u128*>(Ain + e * 256)[q] =
        reinterpret_cast<const u128*>(node_emb + n * HIDDEN)[q];
    const float ic = invc[n];
    U16x8 av;
    av.v = *reinterpret_cast<const u128*>(agg + n * HIDDEN + q * 8);
#pragma unroll
    for (int i = 0; i < 8; ++i)
      Ain[e * 256 + HIDDEN + q * 8 + i] = f32_to_bf16(bf16_to_f32(av.u[i]) * ic);
  }
  __syncthreads();
  { float bv = b1[(wave << 4) + (lane & 15)];
    store_relu_bf16(wmma_rowtile<256>(Ain, WT1, bv, wave, lane), H1, wave, lane); }
  __syncthreads();
  { float bv = b2[(wave << 4) + (lane & 15)];
    store_relu_bf16(wmma_rowtile<HIDDEN>(H1, WT2, bv, wave, lane), H2, wave, lane); }
  __syncthreads();
  { float bv = b3[(wave << 4) + (lane & 15)];
    store_f32(wmma_rowtile<HIDDEN>(H2, WT3, bv, wave, lane), Y, wave, lane); }
  __syncthreads();
  const int r = tid >> 4, j = tid & 15;
  const long long n = tile * 16 + r;
  float mu, rs; ln_stats(&Y[r * HIDDEN], j, mu, rs);
  U16x8 oldv, nw;
  oldv.v = *reinterpret_cast<const u128*>(node_emb + n * HIDDEN + j * 8);
#pragma unroll
  for (int i = 0; i < 8; ++i) {
    int c = j * 8 + i;
    float v = (Y[r * HIDDEN + c] - mu) * rs * g[c] + beta[c] + bf16_to_f32(oldv.u[i]);
    nw.u[i] = f32_to_bf16(v);
  }
  *reinterpret_cast<u128*>(node_emb + n * HIDDEN + j * 8) = nw.v;
}

// ---------------- decoder: 128->128->128->3 (W3 padded to 16 cols) ----------------

__global__ __launch_bounds__(256) void decoder_kernel(
    const unsigned short* __restrict__ node_emb,
    const unsigned short* __restrict__ WT1, const float* __restrict__ b1,
    const unsigned short* __restrict__ WT2, const float* __restrict__ b2,
    const unsigned short* __restrict__ WT3, const float* __restrict__ b3,
    float* __restrict__ out) {
  __shared__ unsigned short Ain[16 * HIDDEN];
  __shared__ unsigned short H1[16 * HIDDEN];
  __shared__ unsigned short H2[16 * HIDDEN];
  const int tid = threadIdx.x, wave = tid >> 5, lane = tid & 31;
  const long long tile = blockIdx.x;
  {
    const int e = tid >> 4, q = tid & 15;
    reinterpret_cast<u128*>(Ain + e * HIDDEN)[q] =
        reinterpret_cast<const u128*>(node_emb + (tile * 16 + e) * HIDDEN)[q];
  }
  __syncthreads();
  { float bv = b1[(wave << 4) + (lane & 15)];
    store_relu_bf16(wmma_rowtile<HIDDEN>(Ain, WT1, bv, wave, lane), H1, wave, lane); }
  __syncthreads();
  { float bv = b2[(wave << 4) + (lane & 15)];
    store_relu_bf16(wmma_rowtile<HIDDEN>(H1, WT2, bv, wave, lane), H2, wave, lane); }
  __syncthreads();
  if (wave == 0) {
    const int colN = lane & 15;
    float bv = (colN < 3) ? b3[colN] : 0.f;
    v8f acc = wmma_rowtile<HIDDEN>(H2, WT3, bv, 0, lane);
    if (colN < 3) {
      const int mbase = (lane >> 4) << 3;
#pragma unroll
      for (int i = 0; i < 8; ++i)
        out[(tile * 16 + mbase + i) * 3 + colN] = acc[i];
    }
  }
}

// ---------------- host launcher ----------------

extern "C" void kernel_launch(void* const* d_in, const int* in_sizes, int n_in,
                              void* d_out, int out_size, void* d_ws, size_t ws_size,
                              hipStream_t stream) {
  (void)in_sizes; (void)n_in; (void)out_size; (void)ws_size;
  const float* node_feat = (const float*)d_in[0];
  const float* edge_feat = (const float*)d_in[1];
  const int*   eidx      = (const int*)d_in[2];
  auto F = [&](int i) { return (const float*)d_in[i]; };

  // Flatten order (jax pytree, dict keys sorted):
  //  0 node_features, 1 edge_features, 2 edge_index,
  //  3..8   decoder   {W1,W2,W3,b1,b2,b3}
  //  9..16  edge_enc  {W1,W2,W3,b1,b2,b3,beta,g}
  // 17..24  node_enc  {W1,W2,W3,b1,b2,b3,beta,g}
  // 25+16s    step s edge_mlp {W1,W2,W3,b1,b2,b3,beta,g}
  // 25+16s+8  step s node_mlp {W1,W2,W3,b1,b2,b3,beta,g}

  char* ws = (char*)d_ws;
  size_t off = 0;
  auto carve = [&](size_t bytes) -> void* {
    void* p = ws + off;
    off += (bytes + 255) & ~(size_t)255;
    return p;
  };
  unsigned short* node_emb = (unsigned short*)carve((size_t)NNODES * HIDDEN * 2);
  unsigned short* edge_emb = (unsigned short*)carve((size_t)NEDGES * HIDDEN * 2);
  unsigned short* agg      = (unsigned short*)carve((size_t)NNODES * HIDDEN * 2);
  float*          invc     = (float*)carve((size_t)NNODES * 4);

  struct WRec { const float* W; int K, N, Kp, Np; unsigned short* WT; };
  WRec recs[70]; int nrec = 0;
  auto addw = [&](int idx, int K, int N, int Kp, int Np) -> unsigned short* {
    unsigned short* wt = (unsigned short*)carve((size_t)Np * Kp * 2);
    recs[nrec++] = { (const float*)d_in[idx], K, N, Kp, Np, wt };
    return wt;
  };

  unsigned short* ne_W1 = addw(17, 10, 128, 32, 128);
  unsigned short* ne_W2 = addw(18, 128, 128, 128, 128);
  unsigned short* ne_W3 = addw(19, 128, 128, 128, 128);
  unsigned short* ee_W1 = addw(9, 9, 128, 32, 128);
  unsigned short* ee_W2 = addw(10, 128, 128, 128, 128);
  unsigned short* ee_W3 = addw(11, 128, 128, 128, 128);
  unsigned short* dc_W1 = addw(3, 128, 128, 128, 128);
  unsigned short* dc_W2 = addw(4, 128, 128, 128, 128);
  unsigned short* dc_W3 = addw(5, 128, 3, 128, 16);  // pad cols 3..15 with zeros

  unsigned short *se_W1[NSTEPS], *se_W2[NSTEPS], *se_W3[NSTEPS];
  unsigned short *sn_W1[NSTEPS], *sn_W2[NSTEPS], *sn_W3[NSTEPS];
  for (int s = 0; s < NSTEPS; ++s) {
    int be = 25 + s * 16, bn = be + 8;
    se_W1[s] = addw(be + 0, 384, 128, 384, 128);
    se_W2[s] = addw(be + 1, 128, 128, 128, 128);
    se_W3[s] = addw(be + 2, 128, 128, 128, 128);
    sn_W1[s] = addw(bn + 0, 256, 128, 256, 128);
    sn_W2[s] = addw(bn + 1, 128, 128, 128, 128);
    sn_W3[s] = addw(bn + 2, 128, 128, 128, 128);
  }

  for (int i = 0; i < nrec; ++i) {
    int total = recs[i].Kp * recs[i].Np;
    wt_conv_kernel<<<(total + 255) / 256, 256, 0, stream>>>(
        recs[i].W, recs[i].WT, recs[i].K, recs[i].N, recs[i].Kp, recs[i].Np);
  }

  // in-degree -> inv_count
  zero_f32_kernel<<<(NNODES + 255) / 256, 256, 0, stream>>>(invc, NNODES);
  count_kernel<<<NEDGES / 256, 256, 0, stream>>>(eidx, invc);
  inv_clip_kernel<<<(NNODES + 255) / 256, 256, 0, stream>>>(invc, NNODES);

  // encoders
  encoder_kernel<<<NNODES / 16, 256, 0, stream>>>(
      node_feat, 10, ne_W1, F(20), ne_W2, F(21), ne_W3, F(22), F(24), F(23), node_emb);
  encoder_kernel<<<NEDGES / 16, 256, 0, stream>>>(
      edge_feat, 9, ee_W1, F(12), ee_W2, F(13), ee_W3, F(14), F(16), F(15), edge_emb);

  // message-passing steps
  for (int s = 0; s < NSTEPS; ++s) {
    int be = 25 + s * 16, bn = be + 8;
    // zero bf16 agg (N*128 bf16 == N*64 f32 words)
    zero_f32_kernel<<<(NNODES * 64) / 256, 256, 0, stream>>>(
        (float*)agg, (long long)NNODES * 64);
    edge_step_kernel<<<NEDGES / 16, 256, 0, stream>>>(
        eidx, node_emb, edge_emb,
        se_W1[s], F(be + 3), se_W2[s], F(be + 4), se_W3[s], F(be + 5),
        F(be + 7), F(be + 6), agg);
    node_step_kernel<<<NNODES / 16, 256, 0, stream>>>(
        agg, invc, node_emb,
        sn_W1[s], F(bn + 3), sn_W2[s], F(bn + 4), sn_W3[s], F(bn + 5),
        F(bn + 7), F(bn + 6));
  }

  decoder_kernel<<<NNODES / 16, 256, 0, stream>>>(
      node_emb, dc_W1, F(6), dc_W2, F(7), dc_W3, F(8), (float*)d_out);
}
